// Encoder_81965155877578
// MI455X (gfx1250) — matrix operand order repaired and checked by reading, not compiled, not run on hardware
//
#include <hip/hip_runtime.h>
#include <hip/hip_bf16.h>

#define HEADS 2
#define CH    128
#define HC    256   // HEADS*CH
#define BN_EPS 1e-5f
#define NEG_SLOPE 0.2f

// LDS staging row stride for B slices (16B pad to spread DS banks)
#define BROW 264    // HC + 8 ushorts

#define USE_ASYNC_LDS 1

typedef __attribute__((ext_vector_type(16))) __bf16       v16bf;
typedef __attribute__((ext_vector_type(8)))  float        v8f;
typedef __attribute__((ext_vector_type(8)))  unsigned int v8u;
typedef __attribute__((ext_vector_type(4)))  unsigned int v4u;

static inline int div_up(long long a, long long b) { return (int)((a + b - 1) / b); }

__device__ __forceinline__ unsigned short f2bf(float f) {
  unsigned u = __float_as_uint(f);
  u += 0x7FFFu + ((u >> 16) & 1u);          // round-to-nearest-even
  return (unsigned short)(u >> 16);
}

__device__ __forceinline__ void atomicMaxF(float* addr, float val) {
  int* ia = (int*)addr;
  int cur = __float_as_int(*addr);
  while (__int_as_float(cur) < val) {
    int old = atomicCAS(ia, cur, __float_as_int(val));
    if (old == cur) break;
    cur = old;
  }
}

__device__ __forceinline__ void async_b128_to_lds(const void* gaddr, void* ldsptr) {
#if USE_ASYNC_LDS
  unsigned lds_off = (unsigned)(size_t)ldsptr;          // AS3 offset lives in low 32 bits
  unsigned long long ga = (unsigned long long)(size_t)gaddr;
  asm volatile("global_load_async_to_lds_b128 %0, %1, off"
               :: "v"(lds_off), "v"(ga) : "memory");
#else
  *(v4u*)ldsptr = *(const v4u*)gaddr;
#endif
}

__device__ __forceinline__ void wait_async_le4() {
#if USE_ASYNC_LDS
#if __has_builtin(__builtin_amdgcn_s_wait_asynccnt)
  __builtin_amdgcn_s_wait_asynccnt(4);
#else
  asm volatile("s_wait_asynccnt 0x4" ::: "memory");
#endif
#endif
}
__device__ __forceinline__ void wait_async_0() {
#if USE_ASYNC_LDS
#if __has_builtin(__builtin_amdgcn_s_wait_asynccnt)
  __builtin_amdgcn_s_wait_asynccnt(0);
#else
  asm volatile("s_wait_asynccnt 0x0" ::: "memory");
#endif
#endif
}

// ---------------- BatchNorm stats (biased, per column) ----------------
__global__ void k_colstats(const float* __restrict__ x, int n, int ld,
                           float* __restrict__ mean, float* __restrict__ rstd) {
  __shared__ float sh[256], sh2[256];
  const int c = blockIdx.x;
  float s = 0.f, s2 = 0.f;
  for (int i = threadIdx.x; i < n; i += blockDim.x) {
    float v = x[(size_t)i * ld + c];
    s += v; s2 += v * v;
  }
  sh[threadIdx.x] = s; sh2[threadIdx.x] = s2;
  __syncthreads();
  for (int o = 128; o > 0; o >>= 1) {
    if (threadIdx.x < o) { sh[threadIdx.x] += sh[threadIdx.x + o]; sh2[threadIdx.x] += sh2[threadIdx.x + o]; }
    __syncthreads();
  }
  if (threadIdx.x == 0) {
    float m = sh[0] / (float)n;
    float v = sh2[0] / (float)n - m * m;
    mean[c] = m;
    rstd[c] = rsqrtf(v + BN_EPS);
  }
}

__global__ void k_bn_apply(const float* __restrict__ x, const float* __restrict__ mean,
                           const float* __restrict__ rstd, const float* __restrict__ g,
                           const float* __restrict__ b, float* __restrict__ out,
                           int total, int ld) {
  int i = blockIdx.x * blockDim.x + threadIdx.x;
  if (i >= total) return;
  int c = i % ld;
  out[i] = (x[i] - mean[c]) * rstd[c] * g[c] + b[c];
}

// ---------------- conv1 tiny projection: [N,5] @ [5,256] (xl and xr fused) ----------------
__global__ void k_lin5(const float* __restrict__ x0,
                       const float* __restrict__ Wl, const float* __restrict__ bl,
                       const float* __restrict__ Wr, const float* __restrict__ br,
                       float* __restrict__ xl, float* __restrict__ xr, int n) {
  __shared__ float feat[16 * 5];
  const int t = threadIdx.x;               // output channel 0..255
  const int node0 = blockIdx.x * 16;
  const int jmax = min(16, n - node0);
  if (t < jmax * 5) feat[t] = x0[(size_t)node0 * 5 + t];
  __syncthreads();
  float wl[5], wr[5];
#pragma unroll
  for (int k = 0; k < 5; ++k) { wl[k] = Wl[k * HC + t]; wr[k] = Wr[k * HC + t]; }
  const float bll = bl[t], brr = br[t];
  for (int j = 0; j < jmax; ++j) {
    float al = bll, ar = brr;
#pragma unroll
    for (int k = 0; k < 5; ++k) { float f = feat[j * 5 + k]; al += f * wl[k]; ar += f * wr[k]; }
    xl[(size_t)(node0 + j) * HC + t] = al;
    xr[(size_t)(node0 + j) * HC + t] = ar;
  }
}

// ---------------- fp32 -> bf16 conversion ----------------
__global__ void k_f2bf(const float* __restrict__ src, unsigned short* __restrict__ dst, int count) {
  int i = blockIdx.x * blockDim.x + threadIdx.x;
  if (i < count) dst[i] = f2bf(src[i]);
}

// ---------------- WMMA GEMM: out[N,256] = bf16(X) @ bf16(W) + bias ----------------
// Block = 256 threads (8 waves) computes 64 rows x 256 cols.
// Wave tile = 32 rows x 64 cols (2 M-tiles x 4 N-tiles, 8 v8f accumulators)
// so each LDS B fragment feeds 2 WMMAs. B K-slices (32x256 bf16 = 16KB)
// double-buffered in LDS via async-to-LDS (ASYNCcnt).
__global__ void __launch_bounds__(256)
k_gemm_bf16(const unsigned short* __restrict__ Xb,
            const unsigned short* __restrict__ Wb,
            const float* __restrict__ bias,
            float* __restrict__ out, int nrows) {
  __shared__ unsigned short Bs[2][32 * BROW];   // 2 x 16.5KB

  const int tid  = threadIdx.x;
  const int wv   = tid >> 5;
  const int lane = tid & 31;
  const int hi   = lane >> 4;
  const int row0 = blockIdx.x * 64;

  const int mbase = row0 + (wv >> 2) * 32;      // wave's 32-row group
  const int colq  = wv & 3;                     // wave's 64-col quarter

  const int m0 = min(mbase + (lane & 15), nrows - 1);
  const int m1 = min(mbase + 16 + (lane & 15), nrows - 1);
  const unsigned short* arow0 = Xb + (size_t)m0 * HC;
  const unsigned short* arow1 = Xb + (size_t)m1 * HC;

  // staging assignment: 8 threads per B row, 64B per thread
  const int srow = tid >> 3;        // 0..31
  const int scol = (tid & 7) * 64;  // byte offset within 512B row

  v8f acc[2][4];
#pragma unroll
  for (int mt = 0; mt < 2; ++mt)
#pragma unroll
    for (int bt = 0; bt < 4; ++bt) acc[mt][bt] = (v8f){0.f,0.f,0.f,0.f,0.f,0.f,0.f,0.f};

  // prologue: stage K-slice 0
  {
    const char* g = (const char*)(Wb + (size_t)srow * HC) + scol;
    char* l = (char*)&Bs[0][srow * BROW] + scol;
#pragma unroll
    for (int j = 0; j < 4; ++j) async_b128_to_lds(g + 16 * j, l + 16 * j);
  }

  for (int ks = 0; ks < 8; ++ks) {
    const int k0 = ks * 32;
    if (ks + 1 < 8) {
      const char* g = (const char*)(Wb + (size_t)(k0 + 32 + srow) * HC) + scol;
      char* l = (char*)&Bs[(ks + 1) & 1][srow * BROW] + scol;
#pragma unroll
      for (int j = 0; j < 4; ++j) async_b128_to_lds(g + 16 * j, l + 16 * j);
      wait_async_le4();     // previous slice complete (in-order), next in flight
    } else {
      wait_async_0();
    }
    __syncthreads();        // all waves staged current slice

    // A fragments: ISA 16-bit 16x32 layout (lane = M row; hi half shifts K +8/+24)
    v8u au0, au1;
#pragma unroll
    for (int p = 0; p < 8; ++p) {
      int kk = k0 + hi * 8 + ((p < 4) ? 2 * p : 16 + 2 * (p - 4));
      au0[p] = *(const unsigned int*)(arow0 + kk);
      au1[p] = *(const unsigned int*)(arow1 + kk);
    }
    v16bf a0 = __builtin_bit_cast(v16bf, au0);
    v16bf a1 = __builtin_bit_cast(v16bf, au1);

    const unsigned short* bbase = &Bs[ks & 1][colq * 64 + (lane & 15)];  // lane = N col
#pragma unroll
    for (int bt = 0; bt < 4; ++bt) {
      v8u bu;
#pragma unroll
      for (int p = 0; p < 8; ++p) {
        int kb = hi * 8 + ((p < 4) ? 2 * p : 16 + 2 * (p - 4));
        bu[p] = (unsigned int)bbase[kb * BROW + bt * 16] | ((unsigned int)bbase[(kb + 1) * BROW + bt * 16] << 16);
      }
      v16bf b = __builtin_bit_cast(v16bf, bu);
      acc[0][bt] = __builtin_amdgcn_wmma_f32_16x16x32_bf16(false, a0, false, b, (short)0, acc[0][bt], false, false);
      acc[1][bt] = __builtin_amdgcn_wmma_f32_16x16x32_bf16(false, a1, false, b, (short)0, acc[1][bt], false, false);
    }
    __syncthreads();        // before next-next slice overwrites this buffer
  }

  // store: ISA 32-bit C/D layout
#pragma unroll
  for (int mt = 0; mt < 2; ++mt) {
#pragma unroll
    for (int bt = 0; bt < 4; ++bt) {
      const int col = colq * 64 + bt * 16 + (lane & 15);
      const float bi = bias[col];
#pragma unroll
      for (int r = 0; r < 8; ++r) {
        int row = mbase + mt * 16 + r + 8 * hi;
        if (row < nrows) out[(size_t)row * HC + col] = acc[mt][bt][r] + bi;
      }
    }
  }
}

// ---------------- graph attention phase ----------------
__global__ void k_init_mden(float* __restrict__ m, float* __restrict__ den, int count) {
  int i = blockIdx.x * blockDim.x + threadIdx.x;
  if (i < count) { m[i] = -INFINITY; den[i] = 0.f; }
}

__global__ void k_zero(float* __restrict__ p, int count) {
  int i = blockIdx.x * blockDim.x + threadIdx.x;
  if (i < count) p[i] = 0.f;
}

// wave per (edge, head): score = att . leaky_relu(xl[src]+xr[dst]+w*We) ; segment-max
__global__ void k_edge_score(const float* __restrict__ xl, const float* __restrict__ xr,
                             const int* __restrict__ src, const int* __restrict__ dst,
                             const float* __restrict__ ew, const float* __restrict__ We,
                             const float* __restrict__ att,
                             float* __restrict__ alpha, float* __restrict__ mbuf, int neh) {
  const int gw = (blockIdx.x * blockDim.x + threadIdx.x) >> 5;
  if (gw >= neh) return;
  const int lane = threadIdx.x & 31;
  const int e = gw >> 1, hh = gw & 1;
  const int s = src[e], d = dst[e];
  const float w = ew[e];
  const float* xlr = xl + (size_t)s * HC + hh * CH;
  const float* xrr = xr + (size_t)d * HC + hh * CH;
  const float* wer = We + hh * CH;
  const float* atr = att + hh * CH;
  float part = 0.f;
#pragma unroll
  for (int c = lane; c < CH; c += 32) {
    float v = xlr[c] + xrr[c] + w * wer[c];
    v = (v > 0.f) ? v : NEG_SLOPE * v;
    part += v * atr[c];
  }
#pragma unroll
  for (int off = 16; off > 0; off >>= 1) part += __shfl_down(part, off, 32);
  if (lane == 0) {
    alpha[gw] = part;
    atomicMaxF(&mbuf[(size_t)d * HEADS + hh], part);
  }
}

__global__ void k_edge_expsum(float* __restrict__ alpha, const int* __restrict__ dst,
                              const float* __restrict__ mbuf, float* __restrict__ den, int neh) {
  int i = blockIdx.x * blockDim.x + threadIdx.x;
  if (i >= neh) return;
  int e = i >> 1, hh = i & 1;
  int d = dst[e];
  float a = __expf(alpha[i] - mbuf[(size_t)d * HEADS + hh]);
  alpha[i] = a;
  atomicAdd(&den[(size_t)d * HEADS + hh], a);
}

// wave per (edge, head): out[dst] += xl[src] * alpha/den
__global__ void k_edge_scatter(const float* __restrict__ xl, const float* __restrict__ alpha,
                               const float* __restrict__ den, const int* __restrict__ src,
                               const int* __restrict__ dst, float* __restrict__ out, int neh) {
  const int gw = (blockIdx.x * blockDim.x + threadIdx.x) >> 5;
  if (gw >= neh) return;
  const int lane = threadIdx.x & 31;
  const int e = gw >> 1, hh = gw & 1;
  const int s = src[e], d = dst[e];
  const float coef = alpha[gw] / (den[(size_t)d * HEADS + hh] + 1e-16f);
  const float* xlr = xl + (size_t)s * HC + hh * CH;
  float* o = out + (size_t)d * HC + hh * CH;
#pragma unroll
  for (int c = lane; c < CH; c += 32) atomicAdd(&o[c], xlr[c] * coef);
}

// ---------------- BN + ReLU + bf16 store (conv1 output) ----------------
__global__ void k_bn_relu_bf16(const float* __restrict__ x, const float* __restrict__ mean,
                               const float* __restrict__ rstd, const float* __restrict__ g,
                               const float* __restrict__ b, unsigned short* __restrict__ xb,
                               int total) {
  int i = blockIdx.x * blockDim.x + threadIdx.x;
  if (i >= total) return;
  int c = i & (HC - 1);
  float v = (x[i] - mean[c]) * rstd[c] * g[c] + b[c];
  xb[i] = f2bf(fmaxf(v, 0.f));
}

// ---------------- head-mean + bias -> final output ----------------
__global__ void k_head_mean(const float* __restrict__ acc, const float* __restrict__ bias,
                            float* __restrict__ out, int total) {
  int i = blockIdx.x * blockDim.x + threadIdx.x;
  if (i >= total) return;
  int nidx = i >> 7, c = i & (CH - 1);
  out[i] = 0.5f * (acc[(size_t)nidx * HC + c] + acc[(size_t)nidx * HC + CH + c]) + bias[c];
}

extern "C" void kernel_launch(void* const* d_in, const int* in_sizes, int n_in,
                              void* d_out, int out_size, void* d_ws, size_t ws_size,
                              hipStream_t stream) {
  const int n = in_sizes[0] / 5;      // 50000
  const int e = in_sizes[2];          // 300000 (edge_weight [E,1])
  const int neh = e * HEADS;

  const float* h     = (const float*)d_in[0];
  const int*   ei    = (const int*)  d_in[1];   // [2,E] (JAX default x64-off -> int32)
  const float* ew    = (const float*)d_in[2];
  const float* bn0_g = (const float*)d_in[3];
  const float* bn0_b = (const float*)d_in[4];
  const float* bn1_g = (const float*)d_in[5];
  const float* bn1_b = (const float*)d_in[6];
  const float* c1_Wl = (const float*)d_in[7];
  const float* c1_bl = (const float*)d_in[8];
  const float* c1_Wr = (const float*)d_in[9];
  const float* c1_br = (const float*)d_in[10];
  const float* c1_We = (const float*)d_in[11];
  const float* c1_att= (const float*)d_in[12];
  /* c1_bias (d_in[13]) absorbed exactly by BN1 */
  const float* mu_Wl = (const float*)d_in[14];
  const float* mu_bl = (const float*)d_in[15];
  const float* mu_Wr = (const float*)d_in[16];
  const float* mu_br = (const float*)d_in[17];
  const float* mu_We = (const float*)d_in[18];
  const float* mu_att= (const float*)d_in[19];
  const float* mu_bias=(const float*)d_in[20];
  const float* ls_Wl = (const float*)d_in[21];
  const float* ls_bl = (const float*)d_in[22];
  const float* ls_Wr = (const float*)d_in[23];
  const float* ls_br = (const float*)d_in[24];
  const float* ls_We = (const float*)d_in[25];
  const float* ls_att= (const float*)d_in[26];
  const float* ls_bias=(const float*)d_in[27];

  const int* src = ei;
  const int* dst = ei + e;
  float* outp = (float*)d_out;

  // ---- workspace carve-up (256B aligned) ----
  char* ws = (char*)d_ws;
  size_t off = 0;
  auto carve = [&](size_t bytes) { char* p = ws + off; off = (off + bytes + 255) & ~(size_t)255; return p; };
  float*          A      = (float*)         carve((size_t)n * HC * 4);   // xl
  float*          B      = (float*)         carve((size_t)n * HC * 4);   // xr, then accumulator
  unsigned short* Xb     = (unsigned short*)carve((size_t)n * HC * 2);   // bf16 activations
  float*          x0     = (float*)         carve((size_t)n * 5 * 4);
  float*          alpha  = (float*)         carve((size_t)neh * 4);
  float*          mbuf   = (float*)         carve((size_t)n * HEADS * 4);
  float*          den    = (float*)         carve((size_t)n * HEADS * 4);
  float*          mean5  = (float*)         carve(5 * 4);
  float*          rstd5  = (float*)         carve(5 * 4);
  float*          mean256= (float*)         carve(HC * 4);
  float*          rstd256= (float*)         carve(HC * 4);
  unsigned short* WbA    = (unsigned short*)carve((size_t)HC * HC * 2);
  unsigned short* WbB    = (unsigned short*)carve((size_t)HC * HC * 2);
  (void)ws_size; (void)n_in; (void)out_size;

  const dim3 blk(256);
  const int wave_blocks = div_up((long long)neh * 32, 256);
  const int eh_blocks   = div_up(neh, 256);
  const int nhc_blocks  = div_up((long long)n * HC, 256);
  const int nh_blocks   = div_up((long long)n * HEADS, 256);
  const int gemm_blocks = div_up(n, 64);

  auto run_edges = [&](const float* xl, float* xr_then_acc, const float* We, const float* att) {
    k_init_mden  <<<nh_blocks,   blk, 0, stream>>>(mbuf, den, n * HEADS);
    k_edge_score <<<wave_blocks, blk, 0, stream>>>(xl, xr_then_acc, src, dst, ew, We, att, alpha, mbuf, neh);
    k_edge_expsum<<<eh_blocks,   blk, 0, stream>>>(alpha, dst, mbuf, den, neh);
    k_zero       <<<nhc_blocks,  blk, 0, stream>>>(xr_then_acc, n * HC);
    k_edge_scatter<<<wave_blocks, blk, 0, stream>>>(xl, alpha, den, src, dst, xr_then_acc, neh);
  };

  // ---- BN0 on h [N,5] ----
  k_colstats<<<5, blk, 0, stream>>>(h, n, 5, mean5, rstd5);
  k_bn_apply<<<div_up((long long)n * 5, 256), blk, 0, stream>>>(h, mean5, rstd5, bn0_g, bn0_b, x0, n * 5, 5);

  // ---- conv1: projections (K=5, scalar), then attention; bias absorbed by BN1 ----
  k_lin5<<<div_up(n, 16), blk, 0, stream>>>(x0, c1_Wl, c1_bl, c1_Wr, c1_br, A, B, n);
  run_edges(A, B, c1_We, c1_att);

  // ---- BN1 + ReLU -> bf16 activations ----
  k_colstats<<<HC, blk, 0, stream>>>(B, n, HC, mean256, rstd256);
  k_bn_relu_bf16<<<nhc_blocks, blk, 0, stream>>>(B, mean256, rstd256, bn1_g, bn1_b, Xb, n * HC);

  // ---- mu layer: WMMA projections, attention, head mean ----
  k_f2bf<<<div_up(HC * HC, 256), blk, 0, stream>>>(mu_Wl, WbA, HC * HC);
  k_f2bf<<<div_up(HC * HC, 256), blk, 0, stream>>>(mu_Wr, WbB, HC * HC);
  k_gemm_bf16<<<gemm_blocks, blk, 0, stream>>>(Xb, WbA, mu_bl, A, n);
  k_gemm_bf16<<<gemm_blocks, blk, 0, stream>>>(Xb, WbB, mu_br, B, n);
  run_edges(A, B, mu_We, mu_att);
  k_head_mean<<<div_up((long long)n * CH, 256), blk, 0, stream>>>(B, mu_bias, outp, n * CH);

  // ---- log_std layer ----
  k_f2bf<<<div_up(HC * HC, 256), blk, 0, stream>>>(ls_Wl, WbA, HC * HC);
  k_f2bf<<<div_up(HC * HC, 256), blk, 0, stream>>>(ls_Wr, WbB, HC * HC);
  k_gemm_bf16<<<gemm_blocks, blk, 0, stream>>>(Xb, WbA, ls_bl, A, n);
  k_gemm_bf16<<<gemm_blocks, blk, 0, stream>>>(Xb, WbB, ls_br, B, n);
  run_edges(A, B, ls_We, ls_att);
  k_head_mean<<<div_up((long long)n * CH, 256), blk, 0, stream>>>(B, ls_bias, outp + (size_t)n * CH, n * CH);
}
